// SelfAtt_28862180229450
// MI455X (gfx1250) — matrix operand order repaired and checked by reading, not compiled, not run on hardware
//
#include <hip/hip_runtime.h>

// ---------------------------------------------------------------------------
// MI455X / gfx1250: wave32, WMMA 16x16x32 bf16 -> f32.
// Problem: B=16, C=512, H=W=32 (N=1024), HEADS=8 (d=64), GROUPS=32.
// All intermediates are bf16 in workspace; with 192MB L2 they stay resident,
// so the QKV GEMM (26 of 36.5 GFLOP) is the compute-bound hot spot: it uses
// 32x64 per-wave tiles (8 wmma per K-step, 8 independent acc chains) with
// double-buffered fragment loads to avoid per-wmma load waits.
// ---------------------------------------------------------------------------

typedef __attribute__((ext_vector_type(16))) __bf16 v16bf;
typedef __attribute__((ext_vector_type(8)))  float  v8f;
typedef __attribute__((ext_vector_type(4)))  unsigned int u32x4;

union Frag16 { v16bf v; u32x4 u[2]; };

static __device__ inline v8f wmma_bf16(v16bf a, v16bf b, v8f c) {
  // 8 args: (neg_a, A, neg_b, B, c_mod, C, reuse_a, reuse_b)
  return __builtin_amdgcn_wmma_f32_16x16x32_bf16(false, a, false, b, (short)0, c,
                                                 false, false);
}

// A-matrix 16x32 bf16 fragment from row-major [row][K], leading dim ld.
// ISA 7.12.2: lane L holds M=L%16; K chunks (k0 + (L/16)*8 + 0..7) and (+16..+23).
static __device__ inline v16bf load_a_frag(const __bf16* base, int rowBase, int ld,
                                           int k0, int lane) {
  int m   = lane & 15;
  int kA0 = k0 + ((lane >> 4) << 3);
  const __bf16* p = base + (size_t)(rowBase + m) * ld + kA0;
  Frag16 f;
  f.u[0] = *(const u32x4*)p;
  f.u[1] = *(const u32x4*)(p + 16);
  return f.v;
}

// B-matrix 32x16 bf16 fragment; storage is [col][K] (col-major matrix), ld = K stride.
// Lane L holds N=L%16; K chunks (k0 + (L/16)*8 + 0..7) and (+16..+23).
static __device__ inline v16bf load_b_frag(const __bf16* base, int colBase, int ld,
                                           int k0, int lane) {
  int n   = lane & 15;
  int kB0 = k0 + ((lane >> 4) << 3);
  const __bf16* p = base + (size_t)(colBase + n) * ld + kB0;
  Frag16 f;
  f.u[0] = *(const u32x4*)p;
  f.u[1] = *(const u32x4*)(p + 16);
  return f.v;
}

// Pack 8 f32 accumulators -> 8 contiguous bf16 (one 16B store).
static __device__ inline void pack_store_bf16x8(__bf16* p, v8f c) {
  union { __bf16 b[8]; u32x4 u; } t;
#pragma unroll
  for (int r = 0; r < 8; ++r) t.b[r] = (__bf16)c[r];
  *(u32x4*)p = t.u;
}

// ---------------------------------------------------------------------------
// K0: f32 -> bf16 weight conversion
// ---------------------------------------------------------------------------
__global__ void convert_f32_bf16(const float* __restrict__ in,
                                 __bf16* __restrict__ out, int n) {
  int i = blockIdx.x * blockDim.x + threadIdx.x;
  if (i < n) out[i] = (__bf16)in[i];
}

// ---------------------------------------------------------------------------
// K1: GroupNorm over (C/G=16, H, W) per (b, g); write xn^T as [b][n][c] bf16.
// ---------------------------------------------------------------------------
__global__ __launch_bounds__(256) void groupnorm_kernel(
    const float* __restrict__ x,      // [B][C][N]
    const float* __restrict__ scale,  // [C]
    const float* __restrict__ bias,   // [C]
    __bf16* __restrict__ xnT) {       // [B][N][C]
  const int C = 512, N = 1024, CG = 16;
  __shared__ float red[256];
  __shared__ float red2[256];
  int b = blockIdx.x >> 5;
  int g = blockIdx.x & 31;
  const float* xb = x + ((size_t)b * C + g * CG) * N;
  int t = threadIdx.x;
  float s = 0.f, ss = 0.f;
  for (int i = t; i < CG * N; i += 256) {
    float v = xb[i];
    s += v; ss += v * v;
  }
  red[t] = s; red2[t] = ss;
  __syncthreads();
  for (int off = 128; off > 0; off >>= 1) {
    if (t < off) { red[t] += red[t + off]; red2[t] += red2[t + off]; }
    __syncthreads();
  }
  float mean = red[0] * (1.0f / (CG * N));
  float var  = red2[0] * (1.0f / (CG * N)) - mean * mean;
  float inv  = rsqrtf(var + 1e-5f);
  for (int i = t; i < CG * N; i += 256) {
    int cl = i >> 10;           // i / N
    int n  = i & 1023;          // i % N
    int c  = g * CG + cl;
    float v = (xb[i] - mean) * inv * scale[c] + bias[c];
    xnT[((size_t)b * N + n) * C + c] = (__bf16)v;
  }
}

// ---------------------------------------------------------------------------
// K2: QKV GEMM. qkv[o][n] = sum_c wqkv[o][c] * xn[c][n], o in [0,1536).
// Block: 128 thr (4 waves); wave tile = 32(o) x 64(n): 2 A-frags x 4 B-frags
// -> 8 wmma per K-step, double-buffered loads. Scatter into q/k/v layouts:
// q^T [b][h][n][64], k,v [b][h][64][n].
// ---------------------------------------------------------------------------
__global__ __launch_bounds__(128) void qkv_gemm_kernel(
    const __bf16* __restrict__ wqkv,  // [1536][512]
    const __bf16* __restrict__ xnT,   // [B][N][C]
    __bf16* __restrict__ qT,          // [B][H][N][64]
    __bf16* __restrict__ kbuf,        // [B][H][64][N]
    __bf16* __restrict__ vbuf) {      // [B][H][64][N]
  const int C = 512, N = 1024;
  int lane = threadIdx.x & 31;
  int wave = threadIdx.x >> 5;
  int b = blockIdx.z;
  int oBase = blockIdx.y * 128 + wave * 32;
  int nBase = blockIdx.x * 64;
  const __bf16* xb = xnT + (size_t)b * N * C;

  v8f acc[2][4] = {};
  v16bf a0 = load_a_frag(wqkv, oBase, C, 0, lane);
  v16bf a1 = load_a_frag(wqkv, oBase + 16, C, 0, lane);
  v16bf bf0 = load_b_frag(xb, nBase +  0, C, 0, lane);
  v16bf bf1 = load_b_frag(xb, nBase + 16, C, 0, lane);
  v16bf bf2 = load_b_frag(xb, nBase + 32, C, 0, lane);
  v16bf bf3 = load_b_frag(xb, nBase + 48, C, 0, lane);

  for (int k0 = 0; k0 < C; k0 += 32) {
    v16bf na0 = a0, na1 = a1, nb0 = bf0, nb1 = bf1, nb2 = bf2, nb3 = bf3;
    int k1 = k0 + 32;
    if (k1 < C) {  // prefetch next K-step before consuming current frags
      na0 = load_a_frag(wqkv, oBase, C, k1, lane);
      na1 = load_a_frag(wqkv, oBase + 16, C, k1, lane);
      nb0 = load_b_frag(xb, nBase +  0, C, k1, lane);
      nb1 = load_b_frag(xb, nBase + 16, C, k1, lane);
      nb2 = load_b_frag(xb, nBase + 32, C, k1, lane);
      nb3 = load_b_frag(xb, nBase + 48, C, k1, lane);
    }
    acc[0][0] = wmma_bf16(a0, bf0, acc[0][0]);
    acc[0][1] = wmma_bf16(a0, bf1, acc[0][1]);
    acc[0][2] = wmma_bf16(a0, bf2, acc[0][2]);
    acc[0][3] = wmma_bf16(a0, bf3, acc[0][3]);
    acc[1][0] = wmma_bf16(a1, bf0, acc[1][0]);
    acc[1][1] = wmma_bf16(a1, bf1, acc[1][1]);
    acc[1][2] = wmma_bf16(a1, bf2, acc[1][2]);
    acc[1][3] = wmma_bf16(a1, bf3, acc[1][3]);
    a0 = na0; a1 = na1; bf0 = nb0; bf1 = nb1; bf2 = nb2; bf3 = nb3;
  }

#pragma unroll
  for (int s = 0; s < 2; ++s) {
    int oW    = oBase + s * 16;
    int third = oW / 512;             // 0=q 1=k 2=v (16-row strips never straddle)
    int oc    = oW & 511;
    int h     = oc >> 6;
    int dW    = oc & 63;              // multiple of 16
    size_t bh = (size_t)b * 8 + h;
    int dOff  = dW + ((lane >> 4) << 3);
#pragma unroll
    for (int t = 0; t < 4; ++t) {
      int n = nBase + t * 16 + (lane & 15);
      if (third == 0) {
        pack_store_bf16x8(qT + (bh * N + n) * 64 + dOff, acc[s][t]);
      } else {
        __bf16* dst = (third == 1 ? kbuf : vbuf) + bh * 64 * (size_t)N;
#pragma unroll
        for (int r = 0; r < 8; ++r)
          dst[(size_t)(dOff + r) * N + n] = (__bf16)acc[s][t][r];
      }
    }
  }
}

// ---------------------------------------------------------------------------
// K3: in-place softmax over n (rows of 1024) on k, bf16 storage / f32 math.
// ---------------------------------------------------------------------------
__global__ __launch_bounds__(256) void softmax_kernel(__bf16* kbuf) {
  const int N = 1024;
  __shared__ float red[256];
  size_t base = (size_t)blockIdx.x * N;
  int t = threadIdx.x;
  float v[4];
  float m = -1e30f;
#pragma unroll
  for (int i = 0; i < 4; ++i) {
    v[i] = (float)kbuf[base + t + i * 256];
    m = fmaxf(m, v[i]);
  }
  red[t] = m;
  __syncthreads();
  for (int off = 128; off > 0; off >>= 1) {
    if (t < off) red[t] = fmaxf(red[t], red[t + off]);
    __syncthreads();
  }
  m = red[0];
  __syncthreads();
  float s = 0.f;
#pragma unroll
  for (int i = 0; i < 4; ++i) { v[i] = __expf(v[i] - m); s += v[i]; }
  red[t] = s;
  __syncthreads();
  for (int off = 128; off > 0; off >>= 1) {
    if (t < off) red[t] += red[t + off];
    __syncthreads();
  }
  float inv = 1.f / red[0];
#pragma unroll
  for (int i = 0; i < 4; ++i)
    kbuf[base + t + i * 256] = (__bf16)(v[i] * inv);
}

// ---------------------------------------------------------------------------
// K4: per (b,h): att[d][e] = sum_n ksm[d][n] v[e][n]  (64x64, K=1024),
// att^T staged in LDS as bf16 [e][d]; then out[e][n] = sum_d attT[e][d] q[d][n]
// (64x1024, K=64). out written as out^T [b][n][c], c = h*64+e.
// ---------------------------------------------------------------------------
__global__ __launch_bounds__(256) void att_kernel(
    const __bf16* __restrict__ ksm,   // [B][H][64][N] (softmaxed)
    const __bf16* __restrict__ vbuf,  // [B][H][64][N]
    const __bf16* __restrict__ qT,    // [B][H][N][64]
    __bf16* __restrict__ outT) {      // [B][N][C]
  const int N = 1024, C = 512;
  __shared__ __bf16 attT[64 * 64];    // [e][d], 8 KB
  int lane = threadIdx.x & 31;
  int wave = threadIdx.x >> 5;
  int b = blockIdx.x >> 3;
  int h = blockIdx.x & 7;
  size_t bh = (size_t)b * 8 + h;
  const __bf16* kb = ksm  + bh * 64 * (size_t)N;
  const __bf16* vb = vbuf + bh * 64 * (size_t)N;
  const __bf16* qb = qT   + bh * (size_t)N * 64;

  // Phase 1: att (4x4 tiles of 16x16), 2 tiles per wave, K-loop over n.
#pragma unroll
  for (int tt = 0; tt < 2; ++tt) {
    int tile = wave * 2 + tt;
    int dT = (tile >> 2) * 16;
    int eT = (tile & 3) * 16;
    v8f acc = {};
    for (int n0 = 0; n0 < N; n0 += 32) {
      v16bf a  = load_a_frag(kb, dT, N, n0, lane);  // rows d, K = n
      v16bf bb = load_b_frag(vb, eT, N, n0, lane);  // cols e, K = n (v^T)
      acc = wmma_bf16(a, bb, acc);
    }
    int e  = eT + (lane & 15);
    int d0 = dT + ((lane >> 4) << 3);
    pack_store_bf16x8(&attT[e * 64 + d0], acc);     // att^T in LDS
  }
  __syncthreads();

  // Phase 2: out[e][n]; 4 eT x 64 nT tiles, round-robin over 8 waves.
  for (int t = wave; t < 4 * 64; t += 8) {
    int eT = (t >> 6) * 16;
    int nT = (t & 63) * 16;
    v8f acc = {};
#pragma unroll
    for (int k0 = 0; k0 < 64; k0 += 32) {
      v16bf a  = load_a_frag(attT, eT, 64, k0, lane);  // LDS A-frag (ds loads)
      v16bf bb = load_b_frag(qb, nT, 64, k0, lane);    // q^T: cols n, K = d
      acc = wmma_bf16(a, bb, acc);
    }
    int n  = nT + (lane & 15);
    int c0 = h * 64 + eT + ((lane >> 4) << 3);
    pack_store_bf16x8(outT + ((size_t)b * N + n) * C + c0, acc);
  }
}

// ---------------------------------------------------------------------------
// K5: projection GEMM + bias: y[b][o][n] = sum_c wproj[o][c]*out[c][n] + pb[o].
// Same 32x64 per-wave tiling with double-buffered fragments as K2.
// ---------------------------------------------------------------------------
__global__ __launch_bounds__(128) void proj_gemm_kernel(
    const __bf16* __restrict__ wproj, // [512][512]
    const __bf16* __restrict__ outT,  // [B][N][C]
    const float* __restrict__ pb,     // [512]
    float* __restrict__ y) {          // [B][512][N]  (f32 output)
  const int C = 512, N = 1024;
  int lane = threadIdx.x & 31;
  int wave = threadIdx.x >> 5;
  int b = blockIdx.z;
  int oBase = blockIdx.y * 128 + wave * 32;
  int nBase = blockIdx.x * 64;
  const __bf16* xb = outT + (size_t)b * N * C;

  v8f acc[2][4] = {};
  v16bf a0 = load_a_frag(wproj, oBase, C, 0, lane);
  v16bf a1 = load_a_frag(wproj, oBase + 16, C, 0, lane);
  v16bf bf0 = load_b_frag(xb, nBase +  0, C, 0, lane);
  v16bf bf1 = load_b_frag(xb, nBase + 16, C, 0, lane);
  v16bf bf2 = load_b_frag(xb, nBase + 32, C, 0, lane);
  v16bf bf3 = load_b_frag(xb, nBase + 48, C, 0, lane);

  for (int k0 = 0; k0 < C; k0 += 32) {
    v16bf na0 = a0, na1 = a1, nb0 = bf0, nb1 = bf1, nb2 = bf2, nb3 = bf3;
    int k1 = k0 + 32;
    if (k1 < C) {
      na0 = load_a_frag(wproj, oBase, C, k1, lane);
      na1 = load_a_frag(wproj, oBase + 16, C, k1, lane);
      nb0 = load_b_frag(xb, nBase +  0, C, k1, lane);
      nb1 = load_b_frag(xb, nBase + 16, C, k1, lane);
      nb2 = load_b_frag(xb, nBase + 32, C, k1, lane);
      nb3 = load_b_frag(xb, nBase + 48, C, k1, lane);
    }
    acc[0][0] = wmma_bf16(a0, bf0, acc[0][0]);
    acc[0][1] = wmma_bf16(a0, bf1, acc[0][1]);
    acc[0][2] = wmma_bf16(a0, bf2, acc[0][2]);
    acc[0][3] = wmma_bf16(a0, bf3, acc[0][3]);
    acc[1][0] = wmma_bf16(a1, bf0, acc[1][0]);
    acc[1][1] = wmma_bf16(a1, bf1, acc[1][1]);
    acc[1][2] = wmma_bf16(a1, bf2, acc[1][2]);
    acc[1][3] = wmma_bf16(a1, bf3, acc[1][3]);
    a0 = na0; a1 = na1; bf0 = nb0; bf1 = nb1; bf2 = nb2; bf3 = nb3;
  }

#pragma unroll
  for (int s = 0; s < 2; ++s) {
    int o0 = oBase + s * 16 + ((lane >> 4) << 3);
#pragma unroll
    for (int t = 0; t < 4; ++t) {
      int n = nBase + t * 16 + (lane & 15);
#pragma unroll
      for (int r = 0; r < 8; ++r) {
        int o = o0 + r;
        y[((size_t)b * C + o) * N + n] = acc[s][t][r] + pb[o];
      }
    }
  }
}

// ---------------------------------------------------------------------------
extern "C" void kernel_launch(void* const* d_in, const int* in_sizes, int n_in,
                              void* d_out, int out_size, void* d_ws, size_t ws_size,
                              hipStream_t stream) {
  const int B = 16, C = 512, N = 1024, HEADS = 8;
  const float* x        = (const float*)d_in[0];
  const float* gn_scale = (const float*)d_in[1];
  const float* gn_bias  = (const float*)d_in[2];
  const float* qkv_w    = (const float*)d_in[3];  // [1536][512]
  const float* proj_w   = (const float*)d_in[4];  // [512][512]
  const float* proj_b   = (const float*)d_in[5];  // [512]
  float* y = (float*)d_out;

  char* ws = (char*)d_ws;
  size_t off = 0;
  auto carve = [&](size_t bytes) {
    char* p = ws + off;
    off = (off + bytes + 255) & ~(size_t)255;
    return p;
  };
  __bf16* wqkv = (__bf16*)carve((size_t)3 * C * C * 2);          // 1.5 MB
  __bf16* wprj = (__bf16*)carve((size_t)C * C * 2);              // 0.5 MB
  __bf16* xnT  = (__bf16*)carve((size_t)B * N * C * 2);          // 16 MB
  __bf16* qT   = (__bf16*)carve((size_t)B * HEADS * N * 64 * 2); // 16 MB
  __bf16* kb   = (__bf16*)carve((size_t)B * HEADS * 64 * N * 2); // 16 MB
  __bf16* vb   = (__bf16*)carve((size_t)B * HEADS * 64 * N * 2); // 16 MB
  __bf16* outT = (__bf16*)carve((size_t)B * N * C * 2);          // 16 MB

  int nqkv = 3 * C * C, nprj = C * C;
  convert_f32_bf16<<<(nqkv + 255) / 256, 256, 0, stream>>>(qkv_w, wqkv, nqkv);
  convert_f32_bf16<<<(nprj + 255) / 256, 256, 0, stream>>>(proj_w, wprj, nprj);

  groupnorm_kernel<<<B * 32, 256, 0, stream>>>(x, gn_scale, gn_bias, xnT);

  // (o,n) tiling: 12 o-groups of 128, 16 n-groups of 64, per batch.
  qkv_gemm_kernel<<<dim3(16, 12, B), 128, 0, stream>>>(wqkv, xnT, qT, kb, vb);

  softmax_kernel<<<B * HEADS * 64, 256, 0, stream>>>(kb);

  att_kernel<<<B * HEADS, 256, 0, stream>>>(kb, vb, qT, outT);

  proj_gemm_kernel<<<dim3(16, 4, B), 128, 0, stream>>>(wprj, outT, proj_b, y);
}
